// MoeDispatcher_39410619908672
// MI455X (gfx1250) — matrix operand order repaired and checked by reading, not compiled, not run on hardware
//
#include <hip/hip_runtime.h>

// ---------------------------------------------------------------------------
// MoE top-2 dispatcher for MI455X (gfx1250, wave32)
//  - bf16 WMMA 16x16x32 fused expert FFN (weights pre-converted, K-contiguous)
//  - TDM gather (tensor_load_to_lds, gather_mode) to stage scattered token
//    rows into LDS, tracked with TENSORcnt
// ---------------------------------------------------------------------------

typedef __attribute__((ext_vector_type(16))) __bf16 v16bf;
typedef __attribute__((ext_vector_type(8)))  __bf16 v8bf;
typedef __attribute__((ext_vector_type(8)))  float  v8f;
typedef __attribute__((ext_vector_type(4)))  unsigned int u32x4;
typedef __attribute__((ext_vector_type(8)))  int i32x8;
typedef __attribute__((ext_vector_type(4)))  int i32x4;

#ifndef __has_builtin
#define __has_builtin(x) 0
#endif
#if __has_builtin(__builtin_amdgcn_tensor_load_to_lds) && \
    __has_builtin(__builtin_amdgcn_s_wait_tensorcnt) &&   \
    __has_builtin(__builtin_amdgcn_readfirstlane)
#define USE_TDM 1
#else
#define USE_TDM 0
#endif

#define B_      4
#define S_      1024
#define D_      1024
#define E_      8
#define F_      4096
#define NTOK    (B_ * S_)          // 4096 tokens
#define TILE_M  16
#define MAX_TILES (NTOK / TILE_M)  // 256 worst-case tiles per expert
#define FC      128                // F-chunk per fused-FFN iteration

// workspace layout (bytes)
#define WS_COUNTS  0u
#define WS_IDS     1024u
#define WS_WTS     (WS_IDS + (unsigned)(E_ * NTOK * 4))
#define WS_W1BF    (524288u)                                  // 512 KiB, 32B aligned
#define WS_W2BF    (WS_W1BF + (unsigned long long)E_ * D_ * F_ * 2ull)

// float -> bf16, round-to-nearest-even via bit manipulation
__device__ __forceinline__ __bf16 f2bf(float f) {
    unsigned u = __builtin_bit_cast(unsigned, f);
    u = (u + 0x7FFFu + ((u >> 16) & 1u)) >> 16;
    unsigned short s = (unsigned short)u;
    return __builtin_bit_cast(__bf16, s);
}

__device__ __forceinline__ float gelu_tanh(float x) {
    // jax.nn.gelu (approximate=True)
    float x3 = x * x * x;
    return 0.5f * x * (1.0f + tanhf(0.7978845608028654f * (x + 0.044715f * x3)));
}

// ---------------------------------------------------------------------------
// 1) zero output + expert counters
// ---------------------------------------------------------------------------
__global__ void k_zero(float* __restrict__ out, int* __restrict__ counts) {
    unsigned i = blockIdx.x * blockDim.x + threadIdx.x;
    if (i < E_) counts[i] = 0;
    if (i < (unsigned)(NTOK * D_)) out[i] = 0.0f;
}

// ---------------------------------------------------------------------------
// 2) top-2 gating + renormalize, build per-expert token lists
// ---------------------------------------------------------------------------
__global__ void k_route(const float* __restrict__ rw, int* __restrict__ counts,
                        int* __restrict__ ids, float* __restrict__ wts) {
    int t = blockIdx.x * blockDim.x + threadIdx.x;
    if (t >= NTOK) return;
    float w[E_];
#pragma unroll
    for (int e = 0; e < E_; ++e) w[e] = rw[t * E_ + e];
    int i1 = 0; float v1 = w[0];
#pragma unroll
    for (int e = 1; e < E_; ++e) if (w[e] > v1) { v1 = w[e]; i1 = e; }
    int i2 = -1; float v2 = -3.4e38f;
#pragma unroll
    for (int e = 0; e < E_; ++e) if (e != i1 && w[e] > v2) { v2 = w[e]; i2 = e; }
    float denom = fmaxf(v1 + v2, 1e-9f);
    int s1 = atomicAdd(&counts[i1], 1);
    ids[i1 * NTOK + s1] = t;  wts[i1 * NTOK + s1] = v1 / denom;
    int s2 = atomicAdd(&counts[i2], 1);
    ids[i2 * NTOK + s2] = t;  wts[i2 * NTOK + s2] = v2 / denom;
}

// ---------------------------------------------------------------------------
// 3) weight conversion fp32 -> bf16, transposed so K is contiguous per column
//    w1bf[e][f][d] = w1[e][d][f]   (GEMM1: K=d, N=f)
//    w2bf[e][d][f] = w2[e][f][d]   (GEMM2: K=f, N=d)
// ---------------------------------------------------------------------------
__global__ void k_conv_w1(const float* __restrict__ w1, __bf16* __restrict__ w1bf) {
    unsigned long long o = (unsigned long long)blockIdx.x * blockDim.x + threadIdx.x;
    unsigned long long e = o / ((unsigned long long)F_ * D_);
    unsigned long long r = o % ((unsigned long long)F_ * D_);
    unsigned long long f = r / D_, d = r % D_;
    w1bf[o] = f2bf(w1[(e * D_ + d) * F_ + f]);
}
__global__ void k_conv_w2(const float* __restrict__ w2, __bf16* __restrict__ w2bf) {
    unsigned long long o = (unsigned long long)blockIdx.x * blockDim.x + threadIdx.x;
    unsigned long long e = o / ((unsigned long long)D_ * F_);
    unsigned long long r = o % ((unsigned long long)D_ * F_);
    unsigned long long d = r / F_, f = r % F_;
    w2bf[o] = f2bf(w2[(e * F_ + f) * D_ + d]);
}

// ---------------------------------------------------------------------------
// 4) fused expert FFN: per block = (expert, 16-token tile), 8 wave32s
//    y[16x1024] = gelu(x[16x1024] @ w1[e]) @ w2[e]  ->  out += gate * (y + b2)
// ---------------------------------------------------------------------------
__global__ void __launch_bounds__(256)
k_moe_ffn(const float* __restrict__ hidden, const float* __restrict__ b1,
          const float* __restrict__ b2, const int* __restrict__ counts,
          const int* __restrict__ ids, const float* __restrict__ wts,
          const __bf16* __restrict__ w1bf, const __bf16* __restrict__ w2bf,
          float* __restrict__ out) {
    __shared__ __align__(16) float  sXF[TILE_M][D_];     // 64 KB raw fp32 gather target
    __shared__ __align__(32) __bf16 sX[TILE_M][D_ + 8];  // 33 KB bf16 A-operand
    __shared__ __align__(32) __bf16 sH[TILE_M][FC + 8];  // 4.3 KB h-chunk
    __shared__ int   sTok[TILE_M];
    __shared__ float sWt[TILE_M];

    const int e    = blockIdx.x / MAX_TILES;
    const int tile = blockIdx.x % MAX_TILES;
    const int cnt  = counts[e];
    if (tile * TILE_M >= cnt) return;   // block-uniform exit

    const int tid  = threadIdx.x;
    const int lane = tid & 31;
    const int wv   = tid >> 5;          // wave 0..7
    const int l15  = lane & 15;
    const int hs   = lane >> 4;         // half-wave select
    const int aK   = hs * 8;            // A-fragment K offset (16-bit layout)
    const int bK   = hs * 16;           // unused; B fragments also load at aK

    if (tid < TILE_M) {
        int slot = tile * TILE_M + tid;
        if (slot < cnt) { sTok[tid] = ids[e * NTOK + slot]; sWt[tid] = wts[e * NTOK + slot]; }
        else            { sTok[tid] = -1;                   sWt[tid] = 0.0f; }
    }
    __syncthreads();

#if USE_TDM
    // ---- stage 16 scattered token rows (fp32) into LDS with one TDM gather ----
    if (wv == 0) {
        const int base = tile * TILE_M;
        unsigned idx[8];
#pragma unroll
        for (int j = 0; j < 8; ++j) {
            int s0 = base + 2 * j, s1 = base + 2 * j + 1;
            unsigned t0 = (s0 < cnt) ? (unsigned)ids[e * NTOK + s0] : 0u;
            unsigned t1 = (s1 < cnt) ? (unsigned)ids[e * NTOK + s1] : 0u;
            idx[j] = (unsigned)__builtin_amdgcn_readfirstlane((int)(t0 | (t1 << 16)));
        }
        unsigned long long ga = (unsigned long long)(uintptr_t)hidden;
        unsigned lds = (unsigned)(uintptr_t)&sXF[0][0];
        lds = (unsigned)__builtin_amdgcn_readfirstlane((int)lds);
        // D# group 0: count=1, gather_mode=1 (16-bit indices), lds addr,
        //             57-bit global addr, type=2 ("image")
        u32x4 g0;
        g0[0] = 1u | (1u << 31);                                     // count | gather
        g0[1] = lds;                                                 // lds_addr
        g0[2] = (unsigned)(ga & 0xFFFFFFFFull);                      // gaddr[31:0]
        g0[3] = (unsigned)((ga >> 32) & 0x01FFFFFFull) | (2u << 30); // gaddr[56:32]|type
        // D# group 1: mask=0, data_size=4B, dims/strides in elements
        i32x8 g1;
        g1[0] = (int)(2u << 16);                                     // data_size=4B
        g1[1] = (int)((D_ & 0xFFFF) << 16);                          // tensor_dim0 lo16
        g1[2] = (int)(((D_ >> 16) & 0xFFFF) | ((NTOK & 0xFFFF) << 16)); // dim0 hi|dim1 lo
        g1[3] = (int)(((NTOK >> 16) & 0xFFFF) | ((D_ & 0xFFFF) << 16)); // dim1 hi|tile_dim0
        g1[4] = (int)(TILE_M & 0xFFFF);                              // tile_dim1 = 16 rows
        g1[5] = (int)D_;                                             // dim0_stride lo32
        g1[6] = 0;                                                   // stride hi / dim1_stride
        g1[7] = 0;
        // D# groups 2+3: 16 x 16-bit gather row indices (token ids)
        i32x4 g2, g3;
#pragma unroll
        for (int j = 0; j < 4; ++j) { g2[j] = (int)idx[j]; g3[j] = (int)idx[4 + j]; }
        // 6-arg form (clang-23 / therock-10.0): extra i32x8 group, then cpol
        i32x8 g4;
#pragma unroll
        for (int j = 0; j < 8; ++j) g4[j] = 0;
        __builtin_amdgcn_tensor_load_to_lds(g0, g1, g2, g3, g4, 0);
        __builtin_amdgcn_s_wait_tensorcnt(0);
    }
    __syncthreads();
#else
    // ---- fallback: manual global -> LDS staging ----
    {
        int m  = tid >> 4;
        int c0 = (tid & 15) * 64;
        int tok = sTok[m];
        const float* src = hidden + (unsigned long long)((tok >= 0) ? tok : 0) * D_ + c0;
#pragma unroll 8
        for (int j = 0; j < 64; ++j) sXF[m][c0 + j] = src[j];
    }
    __syncthreads();
#endif

    // fp32 -> bf16 A-operand (zero padded rows)
    {
        int m  = tid >> 4;
        int c0 = (tid & 15) * 64;
        bool valid = sTok[m] >= 0;
#pragma unroll 8
        for (int j = 0; j < 64; ++j)
            sX[m][c0 + j] = valid ? f2bf(sXF[m][c0 + j]) : f2bf(0.0f);
    }
    __syncthreads();

    const __bf16* w1e = w1bf + (unsigned long long)e * F_ * D_;
    const __bf16* w2e = w2bf + (unsigned long long)e * D_ * F_;

    union AB { v16bf v; v8bf h[2]; };

    v8f yacc[8];   // this wave's 16 x 128 slab of the 16 x 1024 output tile
#pragma unroll
    for (int t = 0; t < 8; ++t)
#pragma unroll
        for (int r = 0; r < 8; ++r) yacc[t][r] = 0.0f;

    for (int fc = 0; fc < F_; fc += FC) {
        // ---- GEMM1: h(16 x 16 per wave) = x(16x1024) @ w1[:, fc+wv*16 ..] ----
        v8f hacc;
#pragma unroll
        for (int r = 0; r < 8; ++r) hacc[r] = 0.0f;
        const int n1 = fc + wv * 16 + l15;                 // F column for this lane
        const __bf16* bcol1 = w1e + (unsigned long long)n1 * D_;  // K contiguous
#pragma unroll 4
        for (int k0 = 0; k0 < D_; k0 += 32) {
            AB a, b;
            a.h[0] = *(const v8bf*)&sX[l15][k0 + aK];
            a.h[1] = *(const v8bf*)&sX[l15][k0 + aK + 16];
            b.h[0] = *(const v8bf*)&bcol1[k0 + aK];
            b.h[1] = *(const v8bf*)&bcol1[k0 + aK + 16];
            hacc = __builtin_amdgcn_wmma_f32_16x16x32_bf16(
                       false, a.v, false, b.v, (short)0, hacc, false, false);
        }
        // ---- bias + gelu -> sH (bf16) ----
        __syncthreads();                       // previous GEMM2 done reading sH
        {
            float bb = b1[(unsigned long long)e * F_ + n1];
#pragma unroll
            for (int r = 0; r < 8; ++r) {
                float g = gelu_tanh(hacc[r] + bb);
                sH[r + hs * 8][wv * 16 + l15] = f2bf(g);
            }
        }
        __syncthreads();
        // ---- GEMM2: yacc += h(16x128) @ w2[fc..fc+127, wv's 128 D-cols] ----
#pragma unroll
        for (int t = 0; t < 8; ++t) {
            const int n2 = wv * 128 + t * 16 + l15;        // D column for this lane
            const __bf16* bcol2 = w2e + (unsigned long long)n2 * F_ + fc;
#pragma unroll
            for (int k0 = 0; k0 < FC; k0 += 32) {
                AB a, b;
                a.h[0] = *(const v8bf*)&sH[l15][k0 + aK];
                a.h[1] = *(const v8bf*)&sH[l15][k0 + aK + 16];
                b.h[0] = *(const v8bf*)&bcol2[k0 + aK];
                b.h[1] = *(const v8bf*)&bcol2[k0 + aK + 16];
                yacc[t] = __builtin_amdgcn_wmma_f32_16x16x32_bf16(
                              false, a.v, false, b.v, (short)0, yacc[t], false, false);
            }
        }
    }

    // ---- epilogue: out[tok, d] += gate * (y + b2[d]) ----
#pragma unroll
    for (int t = 0; t < 8; ++t) {
        int d = wv * 128 + t * 16 + l15;
        float bb = b2[(unsigned long long)e * D_ + d];
#pragma unroll
        for (int r = 0; r < 8; ++r) {
            int m = r + hs * 8;                 // C/D layout: VGPR r -> row r (+8 hi half)
            int tok = sTok[m];
            if (tok >= 0)
                atomicAdd(&out[(unsigned long long)tok * D_ + d],
                          sWt[m] * (yacc[t][r] + bb));
        }
    }
}

// ---------------------------------------------------------------------------
extern "C" void kernel_launch(void* const* d_in, const int* in_sizes, int n_in,
                              void* d_out, int out_size, void* d_ws, size_t ws_size,
                              hipStream_t stream) {
    (void)in_sizes; (void)n_in; (void)out_size; (void)ws_size;
    const float* hidden = (const float*)d_in[0];
    const float* rw     = (const float*)d_in[1];
    const float* w1     = (const float*)d_in[2];
    const float* b1     = (const float*)d_in[3];
    const float* w2     = (const float*)d_in[4];
    const float* b2     = (const float*)d_in[5];
    float* out = (float*)d_out;

    char* ws = (char*)d_ws;
    int*    counts = (int*)(ws + WS_COUNTS);
    int*    ids    = (int*)(ws + WS_IDS);
    float*  wts    = (float*)(ws + WS_WTS);
    __bf16* w1bf   = (__bf16*)(ws + WS_W1BF);
    __bf16* w2bf   = (__bf16*)(ws + WS_W2BF);

    k_zero <<<(NTOK * D_ + 255) / 256, 256, 0, stream>>>(out, counts);
    k_route<<<(NTOK + 255) / 256,      256, 0, stream>>>(rw, counts, ids, wts);

    const unsigned long long wElems = (unsigned long long)E_ * D_ * F_;
    k_conv_w1<<<(unsigned)(wElems / 256), 256, 0, stream>>>(w1, w1bf);
    k_conv_w2<<<(unsigned)(wElems / 256), 256, 0, stream>>>(w2, w2bf);

    k_moe_ffn<<<E_ * MAX_TILES, 256, 0, stream>>>(hidden, b1, b2, counts, ids, wts,
                                                  w1bf, w2bf, out);
}
